// DiscreteContinuousConvS2_27247272526409
// MI455X (gfx1250) — matrix operand order repaired and checked by reading, not compiled, not run on hardware
//
#include <hip/hip_runtime.h>

typedef __attribute__((ext_vector_type(16))) __bf16 v16bf;
typedef __attribute__((ext_vector_type(8)))  float  v8f;

#define B_      2
#define CIN     64
#define COUT    64
#define NLAT    181
#define NLON    360
#define K9      9
#define NNZP    24
#define KTOT    (CIN * K9)        /* 576 */
#define M_      (B_ * NLON)       /* 720 */
#define MT      45                /* ceil(720/16) */
#define NT      4                 /* 64/16 */
#define THREADS 512
#define NWAVE   16

/* x slab: [b][r(3)][c(32)][lon(360)] f32 */
#define XS_FLOATS  (2 * 3 * 32 * NLON)       /* 69120 */
#define XS_BYTES   (XS_FLOATS * 4)           /* 276480 */
#define XS_CHUNK16 (XS_BYTES / 16)           /* 17280 b128 transfers */
#define YCH_BYTES  (M_ * 32 * 2)             /* 46080 bf16 */
#define SMEM_BYTES (XS_BYTES + YCH_BYTES)    /* 322560 <= 327680 */

// ---------------------------------------------------------------------------
// Pre-swizzle weights: [C_OUT][C_IN][9] f32  ->  [f][K = k*64 + c] bf16
// ---------------------------------------------------------------------------
__global__ void disco_reorder_w(const float* __restrict__ w, __bf16* __restrict__ Wb) {
    int idx = blockIdx.x * 256 + threadIdx.x;
    if (idx < COUT * KTOT) {
        int f = idx / KTOT;
        int K = idx - f * KTOT;
        int k = K >> 6;          // K = k*64 + c
        int c = K & 63;
        Wb[idx] = (__bf16)w[(f * CIN + c) * K9 + k];
    }
}

// ---------------------------------------------------------------------------
// Fused DISCO contraction + GEMM. One workgroup per output latitude.
// ---------------------------------------------------------------------------
__global__ __launch_bounds__(THREADS, 1)
void disco_fused(const float* __restrict__ x,
                 const float* __restrict__ psi_vals,
                 const int*   __restrict__ psi_lat_in,
                 const int*   __restrict__ psi_lon_in,
                 const __bf16* __restrict__ Wb,
                 const float* __restrict__ bias,
                 float* __restrict__ out)
{
    extern __shared__ char smem[];
    float*  xsf = (float*)smem;                     // [b][r][c][lon] f32
    __bf16* ych = (__bf16*)(smem + XS_BYTES);       // [m=720][kc=32] bf16

    const int tid  = threadIdx.x;
    const int lat  = blockIdx.x;
    const int wv   = tid >> 5;          // wave id (wave32), also channel-pair id
    const int lane = tid & 31;
    const int h    = lane >> 4;
    const int l16  = lane & 15;

    // LDS byte address of smem base (low 32 bits of flat LDS address = LDS offset)
    const unsigned lds_base = (unsigned)(uintptr_t)(void*)smem;
    const unsigned long long xbase = (unsigned long long)(uintptr_t)x;

    // ---- accumulators: 3 M-tiles x 4 N-tiles per wave ----
    v8f acc[3][NT];
    const v8f vzero = {0.f, 0.f, 0.f, 0.f, 0.f, 0.f, 0.f, 0.f};
#pragma unroll
    for (int i = 0; i < 3; ++i)
#pragma unroll
        for (int nt = 0; nt < NT; ++nt)
            acc[i][nt] = vzero;

    for (int half = 0; half < 2; ++half) {
        const int c0 = half * 32;
        __syncthreads();   // xs reuse across halves

        // ---- async-stage x rows lat-1..lat+1 (clamped), 32 channels, both b ----
        // LDS layout mirrors iteration order: lds offset = i*16.
        for (int i = tid; i < XS_CHUNK16; i += THREADS) {
            int lon4 = i % 90;              // 16-byte chunk within a 1440-B row
            int t    = i / 90;              // (b, r, c)
            int c    = t & 31;
            int t32  = t >> 5;
            int r    = t32 % 3;
            int b    = t32 / 3;
            int li   = lat - 1 + r;
            li = li < 0 ? 0 : (li > NLAT - 1 ? NLAT - 1 : li);
            int gvoff = ((b * CIN + c0 + c) * NLAT + li) * (NLON * 4) + lon4 * 16;
            unsigned ldsa = lds_base + (unsigned)i * 16u;
            asm volatile("global_load_async_to_lds_b128 %0, %1, %2"
                         :
                         : "v"(ldsa), "v"(gvoff), "s"(xbase)
                         : "memory");
        }
        asm volatile("s_wait_asynccnt 0x0" ::: "memory");
        __syncthreads();

        for (int k = 0; k < K9; ++k) {
            // ---- wave-uniform entry data -> scalar registers (s_load path) ----
            float sps[NNZP];
            int   soff[NNZP], sthr[NNZP];
#pragma unroll
            for (int j = 0; j < NNZP; ++j) {
                int e   = (k * NLAT + lat) * NNZP + j;
                sps[j]  = psi_vals[e];
                int ro  = psi_lat_in[e] - lat + 1;         // 0..2 (row clamp handles poles)
                int lon = psi_lon_in[e];
                soff[j] = ro * (32 * NLON) + lon;          // float-index offset within xs[b]
                sthr[j] = NLON - lon;                      // wrap threshold on p
            }

            // ---- roll-contraction: ych[m][c] = sum_j psi * xs[b][roff][c][(lon+p)%360]
            {
                const int cch = 2 * wv;                     // this thread's channel pair
                for (int mi = 0; mi < 23; ++mi) {
                    int m = lane + (mi << 5);
                    if (m < M_) {
                        int b = (m >= NLON) ? 1 : 0;
                        int p = m - b * NLON;
                        int tb = b * (3 * 32 * NLON) + cch * NLON + p;
                        float ax = 0.f, ay = 0.f;
#pragma unroll
                        for (int j = 0; j < NNZP; ++j) {
                            int idx = tb + soff[j];
                            if (p >= sthr[j]) idx -= NLON;  // longitude wrap
                            float v0 = xsf[idx];            // channel cch
                            float v1 = xsf[idx + NLON];     // channel cch+1 (offset:1440)
                            ax = fmaf(sps[j], v0, ax);
                            ay = fmaf(sps[j], v1, ay);
                        }
                        union { __bf16 hv[2]; unsigned u; } pk2;
                        pk2.hv[0] = (__bf16)ax;
                        pk2.hv[1] = (__bf16)ay;
                        *(unsigned*)&ych[m * 32 + cch] = pk2.u;
                    }
                }
            }
            __syncthreads();

            // ---- WMMA: acc += A(ych chunk) x B(weights chunk) ----
            const int K0 = k * 64 + c0;

            v16bf bf[NT];
#pragma unroll
            for (int nt = 0; nt < NT; ++nt) {
                const __bf16* wp = Wb + (size_t)(nt * 16 + l16) * KTOT + K0 + h * 16;
                union { uint4 u[2]; v16bf v; } ub;
                ub.u[0] = *(const uint4*)(wp);
                ub.u[1] = *(const uint4*)(wp + 8);
                bf[nt] = ub.v;
            }

#pragma unroll
            for (int i = 0; i < 3; ++i) {
                int mt = wv + i * NWAVE;
                if (mt < MT) {
                    const char* ap = (const char*)ych + (size_t)(mt * 16 + l16) * 64 + h * 32;
                    union { uint4 u[2]; v16bf v; } ua;
                    ua.u[0] = *(const uint4*)(ap);
                    ua.u[1] = *(const uint4*)(ap + 16);
#pragma unroll
                    for (int nt = 0; nt < NT; ++nt) {
                        acc[i][nt] = __builtin_amdgcn_wmma_f32_16x16x32_bf16(
                            false, ua.v, false, bf[nt], (short)0, acc[i][nt], false, false);
                    }
                }
            }
            __syncthreads();   // ych reused by next k
        }
    }

    // ---- epilogue: D tiles -> out[b][f][lat][p]; acc element d == M = mt*16 + h*8 + d
#pragma unroll
    for (int i = 0; i < 3; ++i) {
        int mt = wv + i * NWAVE;
        if (mt < MT) {
#pragma unroll
            for (int nt = 0; nt < NT; ++nt) {
                int f = nt * 16 + l16;
                float bv = bias[f];
                int Mbase = mt * 16 + h * 8;
                int b = (Mbase >= NLON) ? 1 : 0;
                int p = Mbase - b * NLON;
                float* op = out + (((size_t)(b * COUT + f) * NLAT + lat) * NLON + p);
                v8f c = acc[i][nt];
                float4 lo = make_float4(c[0] + bv, c[1] + bv, c[2] + bv, c[3] + bv);
                float4 hi = make_float4(c[4] + bv, c[5] + bv, c[6] + bv, c[7] + bv);
                *(float4*)(op)     = lo;
                *(float4*)(op + 4) = hi;
            }
        }
    }
}

// ---------------------------------------------------------------------------
extern "C" void kernel_launch(void* const* d_in, const int* in_sizes, int n_in,
                              void* d_out, int out_size, void* d_ws, size_t ws_size,
                              hipStream_t stream) {
    (void)in_sizes; (void)n_in; (void)out_size; (void)ws_size;

    const float* x          = (const float*)d_in[0];
    const float* psi_vals   = (const float*)d_in[1];
    const float* weight     = (const float*)d_in[2];
    const float* bias       = (const float*)d_in[3];
    const int*   psi_lat_in = (const int*)d_in[6];
    const int*   psi_lon_in = (const int*)d_in[7];
    __bf16* Wb  = (__bf16*)d_ws;           // 64*576*2 = 73,728 B of scratch
    float*  out = (float*)d_out;

    disco_reorder_w<<<(COUT * KTOT + 255) / 256, 256, 0, stream>>>(weight, Wb);

    (void)hipFuncSetAttribute((const void*)disco_fused,
                              hipFuncAttributeMaxDynamicSharedMemorySize,
                              (int)SMEM_BYTES);
    disco_fused<<<dim3(NLAT), dim3(THREADS), SMEM_BYTES, stream>>>(
        x, psi_vals, psi_lat_in, psi_lon_in, Wb, bias, out);
}
